// QuanStrust_83829171683954
// MI455X (gfx1250) — matrix operand order repaired and checked by reading, not compiled, not run on hardware
//
#include <hip/hip_runtime.h>

// ---------------------------------------------------------------------------
// Fused:  xq = x@W1^T ; z = circuit(xq+b1+qw) ; out = [z,x]@W2^T + b2
// bf16 WMMA (v_wmma_f32_16x16x32_bf16) with fp32 accumulation on gfx1250.
// Weights pre-converted once to a padded bf16 blob in d_ws, then DMA'd into
// LDS per-workgroup via the Tensor Data Mover (tensor_load_to_lds/TENSORcnt),
// overlapped with fp32->bf16 staging of the x tile.
// ---------------------------------------------------------------------------

typedef __attribute__((ext_vector_type(8)))  float  v8f;
typedef __attribute__((ext_vector_type(16))) __bf16 v16bf;
typedef __attribute__((ext_vector_type(8)))  __bf16 v8bf;
typedef __attribute__((ext_vector_type(4)))  unsigned uint32x4;
typedef __attribute__((ext_vector_type(4)))  int      int32x4;
typedef __attribute__((ext_vector_type(8)))  int      int32x8;

#define TILE_M 128
#define E_DIM  256
#define F_DIM  264
#define YSTR   296   // ushorts per ldsY row (592 B, 16B aligned, bank-spread)
#define WSTR   296   // ushorts per ldsW2 row
#define W1STR  264   // ushorts per ldsW1 row (528 B, 16B aligned)

#define W2_US   (E_DIM * WSTR)          // 75776 ushorts
#define W1_US   (16 * W1STR)            // 4224 ushorts
#define WBLOB_US (W2_US + W1_US)        // 80000 ushorts
#define WBLOB_BYTES (WBLOB_US * 2)      // 160000 bytes
#define WBLOB_U64  (WBLOB_BYTES / 8)    // 20000 8-byte units (TDM data_size=3)

#define SMEM_BYTES ((size_t)(WBLOB_US + TILE_M * YSTR) * 2 + (size_t)TILE_M * 16 * 4)

#if defined(__has_builtin)
#  if __has_builtin(__builtin_amdgcn_tensor_load_to_lds)
#    define HAVE_TDM 1
#  endif
#endif

// fp32 -> bf16 bits, round-to-nearest-even
__device__ __forceinline__ unsigned short f2bf(float f) {
    unsigned u = __builtin_bit_cast(unsigned, f);
    u += 0x7FFFu + ((u >> 16) & 1u);
    return (unsigned short)(u >> 16);
}

// A-fragment (16x32 bf16): half-wave h holds K {kb+8h..+7} and {kb+8h+16..+23}
__device__ __forceinline__ v16bf ldA(const unsigned short* p) {
    v8bf a0 = *(const v8bf*)(p);
    v8bf a1 = *(const v8bf*)(p + 16);
    return __builtin_shufflevector(a0, a1, 0,1,2,3,4,5,6,7,8,9,10,11,12,13,14,15);
}
// B-fragment (32x16 bf16): half-wave h holds 16 contiguous K at kb+16h
__device__ __forceinline__ v16bf ldB(const unsigned short* p) {
    v8bf b0 = *(const v8bf*)(p);
    v8bf b1 = *(const v8bf*)(p + 8);
    return __builtin_shufflevector(b0, b1, 0,1,2,3,4,5,6,7,8,9,10,11,12,13,14,15);
}

// ---- stage x tile -> ldsY cols 8..263 as bf16; zero-pad cols 264..295 ------
__device__ __forceinline__ void stage_x_tile(unsigned short* ldsY,
                                             const float* __restrict__ x,
                                             int row0, int tid) {
    const float4* x4 = (const float4*)(x + (size_t)row0 * E_DIM);
    __builtin_prefetch(x4 + tid, 0, 1);   // global_prefetch_b8
    for (int i = tid; i < TILE_M * 64; i += 256) {
        int r = i >> 6, c4 = (i & 63) << 2;
        float4 v = x4[i];
        unsigned short* p = &ldsY[r * YSTR + 8 + c4];
        p[0] = f2bf(v.x); p[1] = f2bf(v.y); p[2] = f2bf(v.z); p[3] = f2bf(v.w);
    }
    for (int i = tid; i < TILE_M * 32; i += 256) {
        int r = i >> 5, c = 264 + (i & 31);
        ldsY[r * YSTR + c] = 0;
    }
}

// ---- everything after staging: xq WMMA, quantum layer, main WMMA, store ----
__device__ __forceinline__ void compute_all(unsigned short* ldsY,
                                            const unsigned short* ldsW2,
                                            const unsigned short* ldsW1,
                                            float* ldsXQ,
                                            const float* __restrict__ b1,
                                            const float* __restrict__ qw,
                                            const float* __restrict__ b2,
                                            float* __restrict__ out,
                                            int row0, int tid) {
    const int wave = tid >> 5;
    const int lane = tid & 31;
    const int rlo  = lane & 15;
    const int hk   = lane >> 4;

    // xq GEMM: wave w -> 16-row M-tile, N=16 (rows 8..15 of W1 zero), K=256
    {
        v8f acc = {};
        const int r = (wave << 4) + rlo;
        #pragma unroll
        for (int ks = 0; ks < 8; ++ks) {
            const int kb = ks * 32;
            v16bf a = ldB(&ldsY[r * YSTR + 8 + kb + hk * 16]);
            v16bf b = ldB(&ldsW1[rlo * W1STR + kb + hk * 16]);
            acc = __builtin_amdgcn_wmma_f32_16x16x32_bf16(
                      false, a, false, b, (short)0, acc, false, false);
        }
        #pragma unroll
        for (int v = 0; v < 8; ++v) {
            int rr = (wave << 4) + v + (hk << 3);
            ldsXQ[rr * 16 + rlo] = acc[v];
        }
    }
    __syncthreads();

    // quantum layer: theta = xq + b1 + qw ; z via cos + cumprod
    if (tid < TILE_M) {
        float cs[8];
        #pragma unroll
        for (int q = 0; q < 8; ++q) {
            float ang = ldsXQ[tid * 16 + q] + b1[q] + qw[q];
            cs[q] = __cosf(ang);
        }
        float t = cs[1];
        #pragma unroll
        for (int q = 2; q < 8; ++q) t *= cs[q];        // prod c[1..7]
        float z[8];
        z[0] = t;
        float cp = cs[0];
        #pragma unroll
        for (int q = 1; q < 8; ++q) { cp *= cs[q]; z[q] = cp; }
        #pragma unroll
        for (int q = 0; q < 8; ++q) ldsY[tid * YSTR + q] = f2bf(z[q]);
    }
    __syncthreads();

    // main GEMM: out[128 x 256] = y[128 x 288] @ W2bf^T, per-wave N=32
    v8f acc[8][2] = {};
    const int eb = wave * 32;
    #pragma unroll
    for (int ks = 0; ks < 9; ++ks) {
        const int kb = ks * 32;
        v16bf bf0 = ldB(&ldsW2[(eb +      rlo) * WSTR + kb + hk * 16]);
        v16bf bf1 = ldB(&ldsW2[(eb + 16 + rlo) * WSTR + kb + hk * 16]);
        #pragma unroll
        for (int mt = 0; mt < 8; ++mt) {
            v16bf a = ldB(&ldsY[(mt * 16 + rlo) * YSTR + kb + hk * 16]);
            acc[mt][0] = __builtin_amdgcn_wmma_f32_16x16x32_bf16(
                             false, a, false, bf0, (short)0, acc[mt][0], false, false);
            acc[mt][1] = __builtin_amdgcn_wmma_f32_16x16x32_bf16(
                             false, a, false, bf1, (short)0, acc[mt][1], false, false);
        }
    }

    // epilogue: + b2, fp32 store
    const float bv0 = b2[eb + rlo];
    const float bv1 = b2[eb + 16 + rlo];
    #pragma unroll
    for (int mt = 0; mt < 8; ++mt) {
        #pragma unroll
        for (int v = 0; v < 8; ++v) {
            const size_t m = (size_t)(row0 + mt * 16 + v + (hk << 3));
            out[m * E_DIM + eb +      rlo] = acc[mt][0][v] + bv0;
            out[m * E_DIM + eb + 16 + rlo] = acc[mt][1][v] + bv1;
        }
    }
}

// ---- prep kernel: build padded bf16 weight blob in d_ws once ---------------
// layout: [ W2bf 256 x 296 | W1bf 16 x 264 ]  (matches LDS layout exactly)
__global__ __launch_bounds__(256)
void quanstrust_prep(const float* __restrict__ W1,
                     const float* __restrict__ W2,
                     unsigned short* __restrict__ ws) {
    int i = blockIdx.x * 256 + threadIdx.x;
    if (i < W2_US) {
        int e = i / WSTR, f = i - e * WSTR;
        ws[i] = f2bf((f < F_DIM) ? W2[e * F_DIM + f] : 0.0f);
    } else if (i < WBLOB_US) {
        int j = i - W2_US;
        int q = j / W1STR, k = j - q * W1STR;
        ws[i] = f2bf((q < 8 && k < 256) ? W1[q * 256 + k] : 0.0f);
    }
}

// ---- main kernel: TDM weight blob -> LDS, overlapped with x staging --------
__global__ __launch_bounds__(256)
void quanstrust_main(const float* __restrict__ x,
                     const float* __restrict__ b1,
                     const float* __restrict__ qw,
                     const unsigned short* __restrict__ wblob,
                     const float* __restrict__ b2,
                     float* __restrict__ out) {
    extern __shared__ __align__(16) unsigned short smem[];
    unsigned short* ldsW2 = smem;                         // offset 0 (TDM dest)
    unsigned short* ldsW1 = smem + W2_US;
    unsigned short* ldsY  = smem + WBLOB_US;
    float*          ldsXQ = (float*)(smem + WBLOB_US + TILE_M * YSTR);

    const int tid  = threadIdx.x;
    const int row0 = blockIdx.x * TILE_M;

#if defined(HAVE_TDM)
    if (tid < 32) {   // wave 0 issues the DMA; TDM ignores EXEC
        unsigned long long ga = (unsigned long long)(uintptr_t)wblob;
        uint32x4 g0;
        g0[0] = 1u;                                   // count=1, user mode
        g0[1] = 0u;                                   // lds_addr = 0 (blob at base)
        g0[2] = (unsigned)(ga & 0xFFFFFFFFu);         // global_addr[31:0]
        g0[3] = (unsigned)((ga >> 32) & 0x01FFFFFFu)  // global_addr[56:32]
              | (2u << 30);                           // type = 2 ("image")
        int32x8 g1;
        g1[0] = 0x00030000;                           // data_size=3 (8B units)
        g1[1] = (int)((WBLOB_U64 & 0xFFFFu) << 16);   // tensor_dim0[15:0]
        g1[2] = (int)((WBLOB_U64 >> 16) & 0xFFFFu)    // tensor_dim0[31:16]
              | (1 << 16);                            // tensor_dim1 = 1
        g1[3] = (int)((WBLOB_U64 & 0xFFFFu) << 16);   // tile_dim0 = 20000 units
        g1[4] = 0;                                    // tile_dim1/2 unused
        g1[5] = (int)WBLOB_U64;                       // tensor_dim0_stride
        g1[6] = 0;
        g1[7] = 0;
        int32x4 gz = {};
#if __clang_major__ >= 23
        int32x8 gz8 = {};
        __builtin_amdgcn_tensor_load_to_lds(g0, g1, gz, gz, gz8, 0);
#else
        __builtin_amdgcn_tensor_load_to_lds(g0, g1, gz, gz, 0);
#endif
    }
    stage_x_tile(ldsY, x, row0, tid);                 // overlaps the TDM DMA
    if (tid < 32) __builtin_amdgcn_s_wait_tensorcnt(0);
#else
    // fallback: cooperative raw 8-byte copies of the prepared blob
    {
        const unsigned long long* src = (const unsigned long long*)wblob;
        unsigned long long* dst = (unsigned long long*)smem;
        for (int i = tid; i < WBLOB_U64; i += 256) dst[i] = src[i];
    }
    stage_x_tile(ldsY, x, row0, tid);
#endif
    __syncthreads();

    compute_all(ldsY, ldsW2, ldsW1, ldsXQ, b1, qw, b2, out, row0, tid);
}

// ---- standalone fallback (no workspace): convert weights in-kernel ---------
__global__ __launch_bounds__(256)
void quanstrust_standalone(const float* __restrict__ x,
                           const float* __restrict__ W1,
                           const float* __restrict__ b1,
                           const float* __restrict__ qw,
                           const float* __restrict__ W2,
                           const float* __restrict__ b2,
                           float* __restrict__ out) {
    extern __shared__ __align__(16) unsigned short smem[];
    unsigned short* ldsW2 = smem;
    unsigned short* ldsW1 = smem + W2_US;
    unsigned short* ldsY  = smem + WBLOB_US;
    float*          ldsXQ = (float*)(smem + WBLOB_US + TILE_M * YSTR);

    const int tid  = threadIdx.x;
    const int row0 = blockIdx.x * TILE_M;

    for (int i = tid; i < W2_US; i += 256) {
        int e = i / WSTR, f = i - e * WSTR;
        ldsW2[i] = f2bf((f < F_DIM) ? W2[e * F_DIM + f] : 0.0f);
    }
    for (int i = tid; i < W1_US; i += 256) {
        int q = i / W1STR, k = i - q * W1STR;
        ldsW1[i] = f2bf((q < 8 && k < 256) ? W1[q * 256 + k] : 0.0f);
    }
    stage_x_tile(ldsY, x, row0, tid);
    __syncthreads();

    compute_all(ldsY, ldsW2, ldsW1, ldsXQ, b1, qw, b2, out, row0, tid);
}

extern "C" void kernel_launch(void* const* d_in, const int* in_sizes, int n_in,
                              void* d_out, int out_size, void* d_ws, size_t ws_size,
                              hipStream_t stream) {
    (void)in_sizes; (void)n_in; (void)out_size;
    const float* x  = (const float*)d_in[0];
    const float* W1 = (const float*)d_in[1];
    const float* b1 = (const float*)d_in[2];
    const float* qw = (const float*)d_in[3];
    const float* W2 = (const float*)d_in[4];
    const float* b2 = (const float*)d_in[5];
    float* out = (float*)d_out;

    (void)hipFuncSetAttribute((const void*)quanstrust_main,
                              hipFuncAttributeMaxDynamicSharedMemorySize,
                              (int)SMEM_BYTES);
    (void)hipFuncSetAttribute((const void*)quanstrust_standalone,
                              hipFuncAttributeMaxDynamicSharedMemorySize,
                              (int)SMEM_BYTES);

    const int total_rows = 8 * 8192;            // B * S = 65536
    dim3 grid(total_rows / TILE_M);             // 512 workgroups
    dim3 block(256);                            // 8 wave32

    if (d_ws != nullptr && ws_size >= (size_t)WBLOB_BYTES) {
        quanstrust_prep<<<(WBLOB_US + 255) / 256, 256, 0, stream>>>(
            W1, W2, (unsigned short*)d_ws);
        quanstrust_main<<<grid, block, SMEM_BYTES, stream>>>(
            x, b1, qw, (const unsigned short*)d_ws, b2, out);
    } else {
        quanstrust_standalone<<<grid, block, SMEM_BYTES, stream>>>(
            x, W1, b1, qw, W2, b2, out);
    }
}
